// GatedDeltaNetConv2d_37417755083572
// MI455X (gfx1250) — matrix operand order repaired and checked by reading, not compiled, not run on hardware
//
#include <hip/hip_runtime.h>
#include <hip/hip_bf16.h>

// ---------------------------------------------------------------------------
// Gated DeltaNet w/ causal dwconv — MI455X (gfx1250, wave32, WMMA) pipeline
//   B=2, T=2048, H=2048, HK=16, DK=128, HV=32, DV=128, VE=256, K=4
// ---------------------------------------------------------------------------

typedef __bf16 bf16;
typedef __attribute__((ext_vector_type(16))) __bf16 v16bf;
typedef __attribute__((ext_vector_type(8)))  float  v8f;
typedef __attribute__((ext_vector_type(4)))  unsigned int u32x4;

#define BB   2
#define TT   2048
#define HH   2048
#define NTOK (BB * TT)      // 4096 tokens
#define HKc  16
#define DKc  128
#define VEc  256
#define CQ   2048           // HK*DK
#define CV   4096           // HV*DV == HK*VE
#define KW   4              // conv taps

// ---------------------------------------------------------------------------
// f32 -> bf16 elementwise convert
// ---------------------------------------------------------------------------
__global__ __launch_bounds__(256) void cvt_f32_bf16(const float* __restrict__ s,
                                                    bf16* __restrict__ d, size_t n) {
  size_t i = (size_t)blockIdx.x * 256 + threadIdx.x;
  if (i < n) d[i] = (bf16)s[i];
}

// ---------------------------------------------------------------------------
// Tiled bf16 WMMA GEMM:  C[M,NT] = A[M,K] * X^T   (X is [NT,K] row-major)
//   Block tile 128M x 64N, 256 threads (8 waves).
//   Wave w owns a 2x2 grid of 16x16 C tiles: mb = (w&3)*2, nb = (w>>2)*2
//   -> 4 WMMA per 32-K step for 8 ds_load_b128 (2 LDS loads / WMMA, frags
//      reused across two WMMAs each in registers).
//   LDS pitch = 40 halves (80B): staged rows and fragment reads are all 16B
//   aligned; 80B row stride rotates banks.
// Fragment layouts follow CDNA5 ISA 7.12.2:
//   A 16x32: lane l (m = l%16, g = l/16) halves 0..7 = k[g*8 .. g*8+7],
//            halves 8..15 = k[16+g*8 .. 16+g*8+7]   (two 16B runs)
//   B 32x16: lane l (n = l%16, g = l/16) same k order as A (two 16B runs)
//   C/D f32: lane l, vgpr r -> n = l%16, m = r + 8*(l/16)
// ---------------------------------------------------------------------------
union FragAB { v16bf v; u32x4 u[2]; };

#define WMMA_BF16(Afrag, Bfrag, Cacc)                                          \
  __builtin_amdgcn_wmma_f32_16x16x32_bf16(false, (Afrag).v, false, (Bfrag).v,  \
                                          (short)0, (Cacc), false, false)

template <bool TOKEN_MAJOR>
__global__ __launch_bounds__(256)
void gemm_bf16_wmma(const bf16* __restrict__ A, const bf16* __restrict__ X,
                    float* __restrict__ C, int M, int K, int NT) {
  __shared__ bf16 Als[128 * 40];
  __shared__ bf16 Bls[64 * 40];

  const int tid  = threadIdx.x;
  const int lane = tid & 31;
  const int wv   = tid >> 5;        // 0..7
  const int m0   = blockIdx.y * 128;
  const int n0   = blockIdx.x * 64;
  const int mb   = (wv & 3) * 2;    // two 16-row blocks: mb, mb+1
  const int nb   = (wv >> 2) * 2;   // two 16-col blocks: nb, nb+1

  const int l16 = lane & 15;
  const int hg  = lane >> 4;        // lane group 0/1

  // staging: A rows 0..127 (2 b128 chunks/thread), B rows 0..63 (1 chunk)
  const int sr = tid >> 2;          // 0..63
  const int sc = (tid & 3) * 8;     // half offset within 32-wide row

  v8f acc00 = {}, acc01 = {}, acc10 = {}, acc11 = {};

  const int nk = K / 32;
  for (int kk = 0; kk < nk; ++kk) {
    const int k0 = kk * 32;
    *(u32x4*)(Als + sr * 40 + sc) =
        *(const u32x4*)(A + (size_t)(m0 + sr) * K + k0 + sc);
    *(u32x4*)(Als + (64 + sr) * 40 + sc) =
        *(const u32x4*)(A + (size_t)(m0 + 64 + sr) * K + k0 + sc);
    *(u32x4*)(Bls + sr * 40 + sc) =
        *(const u32x4*)(X + (size_t)(n0 + sr) * K + k0 + sc);
    if (kk + 1 < nk) {  // global_prefetch_b8 next K tile
      __builtin_prefetch(A + (size_t)(m0 + sr) * K + k0 + 32 + sc, 0, 1);
      __builtin_prefetch(A + (size_t)(m0 + 64 + sr) * K + k0 + 32 + sc, 0, 1);
      __builtin_prefetch(X + (size_t)(n0 + sr) * K + k0 + 32 + sc, 0, 1);
    }
    __syncthreads();

    FragAB a0, a1, b0, b1;
    const bf16* ap0 = Als + (mb * 16 + l16) * 40 + hg * 8;
    const bf16* ap1 = Als + ((mb + 1) * 16 + l16) * 40 + hg * 8;
    a0.u[0] = *(const u32x4*)(ap0);
    a0.u[1] = *(const u32x4*)(ap0 + 16);
    a1.u[0] = *(const u32x4*)(ap1);
    a1.u[1] = *(const u32x4*)(ap1 + 16);
    const bf16* bp0 = Bls + (nb * 16 + l16) * 40 + hg * 8;
    const bf16* bp1 = Bls + ((nb + 1) * 16 + l16) * 40 + hg * 8;
    b0.u[0] = *(const u32x4*)(bp0);
    b0.u[1] = *(const u32x4*)(bp0 + 16);
    b1.u[0] = *(const u32x4*)(bp1);
    b1.u[1] = *(const u32x4*)(bp1 + 16);

    acc00 = WMMA_BF16(a0, b0, acc00);
    acc01 = WMMA_BF16(a0, b1, acc01);
    acc10 = WMMA_BF16(a1, b0, acc10);
    acc11 = WMMA_BF16(a1, b1, acc11);
    __syncthreads();
  }

#pragma unroll
  for (int r = 0; r < 8; ++r) {
    const int mA = m0 + mb * 16 + r + hg * 8;
    const int mB = mA + 16;
    const int nA = n0 + nb * 16 + l16;
    const int nB = nA + 16;
    if (TOKEN_MAJOR) {               // C[n][m]  (final out [token, H])
      C[(size_t)nA * M + mA] = acc00[r];
      C[(size_t)nB * M + mA] = acc01[r];
      C[(size_t)nA * M + mB] = acc10[r];
      C[(size_t)nB * M + mB] = acc11[r];
    } else {                         // C[m][n]  (pre-act [channel, token])
      C[(size_t)mA * NT + nA] = acc00[r];
      C[(size_t)mA * NT + nB] = acc01[r];
      C[(size_t)mB * NT + nA] = acc10[r];
      C[(size_t)mB * NT + nB] = acc11[r];
    }
  }
}

// ---------------------------------------------------------------------------
// Causal depthwise conv (K=4) + bias + SiLU, with [c][n] -> [n][c] transpose
// ---------------------------------------------------------------------------
__global__ __launch_bounds__(256)
void conv_silu_tr(const float* __restrict__ pre, const float* __restrict__ w,
                  const float* __restrict__ bias, float* __restrict__ act,
                  int Cc) {
  size_t idx = (size_t)blockIdx.x * 256 + threadIdx.x;
  if (idx >= (size_t)Cc * NTOK) return;
  const int c = (int)(idx / NTOK);
  const int n = (int)(idx % NTOK);
  const int t = n & (TT - 1);            // position within batch
  float y = 0.f;
#pragma unroll
  for (int j = 0; j < KW; ++j) {
    const int tj = t - (KW - 1) + j;
    if (tj >= 0) y = fmaf(w[c * KW + j], pre[(size_t)c * NTOK + (n - (KW - 1) + j)], y);
  }
  const float z = y + bias[c];
  act[(size_t)n * Cc + c] = z / (1.f + __expf(-z));   // SiLU
}

// ---------------------------------------------------------------------------
// alpha/beta gates: sigmoid(Wa x), sigmoid(Wb x) — 32 dots of length H / token
// ---------------------------------------------------------------------------
__global__ __launch_bounds__(256)
void gates_kernel(const float* __restrict__ x, const float* __restrict__ Wa,
                  const float* __restrict__ Wb, float* __restrict__ al,
                  float* __restrict__ be) {
  __shared__ float red[256];
  const int n   = blockIdx.x;
  const int tid = threadIdx.x;
  const int row = tid >> 3;   // 0..31 (0-15 -> Wa, 16-31 -> Wb)
  const int sub = tid & 7;
  const float* W  = (row < HKc) ? (Wa + (size_t)row * HH)
                                : (Wb + (size_t)(row - HKc) * HH);
  const float* xr = x + (size_t)n * HH;
  float s = 0.f;
  for (int h = sub; h < HH; h += 8) s = fmaf(W[h], xr[h], s);
  red[tid] = s;
  __syncthreads();
  if (sub == 0) {
    float tot = 0.f;
#pragma unroll
    for (int j = 0; j < 8; ++j) tot += red[(row << 3) + j];
    const float sg = 1.f / (1.f + __expf(-tot));
    if (row < HKc) al[(size_t)n * HKc + row]         = sg;
    else           be[(size_t)n * HKc + (row - HKc)] = sg;
  }
}

// ---------------------------------------------------------------------------
// L2 normalize each 128-vector in place (vectors tile the array contiguously)
// ---------------------------------------------------------------------------
__global__ __launch_bounds__(128)
void l2norm_kernel(float* __restrict__ a) {
  __shared__ float red[128];
  const size_t base = (size_t)blockIdx.x * 128;
  const float v = a[base + threadIdx.x];
  red[threadIdx.x] = v * v;
  __syncthreads();
  for (int s = 64; s > 0; s >>= 1) {
    if (threadIdx.x < s) red[threadIdx.x] += red[threadIdx.x + s];
    __syncthreads();
  }
  a[base + threadIdx.x] = v * rsqrtf(red[0] + 1e-12f);
}

// ---------------------------------------------------------------------------
// Serial delta-rule recurrence. One workgroup per (b, h); thread v owns state
// column S[:, v] (128 f32) in registers — no cross-lane reductions needed.
// Emits bf16 outs for the Wo WMMA GEMM and the final f32 state.
// ---------------------------------------------------------------------------
__global__ __launch_bounds__(256)
void recurrence_kernel(const float* __restrict__ actQ, const float* __restrict__ actK,
                       const float* __restrict__ actV, const float* __restrict__ al,
                       const float* __restrict__ be, const float* __restrict__ s0,
                       bf16* __restrict__ outs, float* __restrict__ stateOut) {
  const int b = blockIdx.x >> 4;
  const int h = blockIdx.x & 15;
  const int v = threadIdx.x;                       // 0..255 = VE column
  __shared__ float kq[256];                        // [0..127]=k, [128..255]=q
  __shared__ float vv[256];

  float s[DKc];
  const size_t sbase = ((size_t)(b * HKc + h) * DKc) * VEc;
#pragma unroll
  for (int d = 0; d < DKc; ++d) s[d] = s0[sbase + (size_t)d * VEc + v];

  for (int t = 0; t < TT; ++t) {
    const size_t n = (size_t)b * TT + t;
    if (v < DKc) kq[v] = actK[n * CQ + h * DKc + v];
    else         kq[v] = actQ[n * CQ + h * DKc + (v - DKc)];
    vv[v] = actV[n * CV + h * VEc + v];
    __syncthreads();

    const float a  = al[n * HKc + h];
    const float bg = be[n * HKc + h];

    float retr = 0.f;
#pragma unroll
    for (int d = 0; d < DKc; ++d) {                // S *= alpha; retr = S^T k
      s[d] *= a;
      retr = fmaf(s[d], kq[d], retr);
    }
    const float werr = bg * (vv[v] - retr);        // beta * (v - retrieved)
    float o = 0.f;
#pragma unroll
    for (int d = 0; d < DKc; ++d) {                // S += k (x) werr; o = S^T q
      s[d] = fmaf(kq[d], werr, s[d]);
      o = fmaf(s[d], kq[DKc + d], o);
    }
    outs[n * CV + h * VEc + v] = (bf16)o;
    __syncthreads();
  }
#pragma unroll
  for (int d = 0; d < DKc; ++d) stateOut[sbase + (size_t)d * VEc + v] = s[d];
}

// ---------------------------------------------------------------------------
// Host launcher
// ---------------------------------------------------------------------------
extern "C" void kernel_launch(void* const* d_in, const int* in_sizes, int n_in,
                              void* d_out, int out_size, void* d_ws, size_t ws_size,
                              hipStream_t stream) {
  const float* x     = (const float*)d_in[0];
  const float* state = (const float*)d_in[1];
  const float* Wq    = (const float*)d_in[2];
  const float* Wk    = (const float*)d_in[3];
  const float* Wv    = (const float*)d_in[4];
  const float* Wo    = (const float*)d_in[5];
  const float* Wa    = (const float*)d_in[6];
  const float* Wb    = (const float*)d_in[7];
  const float* qcW   = (const float*)d_in[8];
  const float* qcB   = (const float*)d_in[9];
  const float* kcW   = (const float*)d_in[10];
  const float* kcB   = (const float*)d_in[11];
  const float* vcW   = (const float*)d_in[12];
  const float* vcB   = (const float*)d_in[13];

  float* outF   = (float*)d_out;                       // [NTOK, H]
  float* stateF = outF + (size_t)NTOK * HH;            // [B, HK, DK, VE]

  char* ws = (char*)d_ws;
  size_t off = 0;
  auto alloc = [&](size_t bytes) -> void* {
    void* p = ws + off;
    off = (off + bytes + 255) & ~(size_t)255;
    return p;
  };

  bf16*  x_bf  = (bf16*)alloc((size_t)NTOK * HH * 2);
  bf16*  wq_bf = (bf16*)alloc((size_t)CQ * HH * 2);
  bf16*  wk_bf = (bf16*)alloc((size_t)CQ * HH * 2);
  bf16*  wv_bf = (bf16*)alloc((size_t)CV * HH * 2);
  bf16*  wo_bf = (bf16*)alloc((size_t)HH * CV * 2);
  float* preQ  = (float*)alloc((size_t)CQ * NTOK * 4);
  float* preK  = (float*)alloc((size_t)CQ * NTOK * 4);
  float* preV  = (float*)alloc((size_t)CV * NTOK * 4);
  float* actQ  = (float*)alloc((size_t)NTOK * CQ * 4);
  float* actK  = (float*)alloc((size_t)NTOK * CQ * 4);
  float* actV  = (float*)alloc((size_t)NTOK * CV * 4);
  float* alB   = (float*)alloc((size_t)NTOK * HKc * 4);
  float* beB   = (float*)alloc((size_t)NTOK * HKc * 4);
  bf16*  outs_bf = (bf16*)preQ;   // alias: preQ dead after conv stage (same size)

  // 1) down-convert activations + weights to bf16
  {
    size_t n;
    n = (size_t)NTOK * HH;
    cvt_f32_bf16<<<dim3((unsigned)((n + 255) / 256)), 256, 0, stream>>>(x, x_bf, n);
    n = (size_t)CQ * HH;
    cvt_f32_bf16<<<dim3((unsigned)((n + 255) / 256)), 256, 0, stream>>>(Wq, wq_bf, n);
    cvt_f32_bf16<<<dim3((unsigned)((n + 255) / 256)), 256, 0, stream>>>(Wk, wk_bf, n);
    n = (size_t)CV * HH;
    cvt_f32_bf16<<<dim3((unsigned)((n + 255) / 256)), 256, 0, stream>>>(Wv, wv_bf, n);
    n = (size_t)HH * CV;
    cvt_f32_bf16<<<dim3((unsigned)((n + 255) / 256)), 256, 0, stream>>>(Wo, wo_bf, n);
  }

  // 2) projection GEMMs (WMMA bf16): pre[c][token]
  gemm_bf16_wmma<false><<<dim3(NTOK / 64, CQ / 128), 256, 0, stream>>>(
      wq_bf, x_bf, preQ, CQ, HH, NTOK);
  gemm_bf16_wmma<false><<<dim3(NTOK / 64, CQ / 128), 256, 0, stream>>>(
      wk_bf, x_bf, preK, CQ, HH, NTOK);
  gemm_bf16_wmma<false><<<dim3(NTOK / 64, CV / 128), 256, 0, stream>>>(
      wv_bf, x_bf, preV, CV, HH, NTOK);

  // 3) causal dwconv + SiLU, transpose to [token][c]
  {
    size_t n = (size_t)CQ * NTOK;
    conv_silu_tr<<<dim3((unsigned)((n + 255) / 256)), 256, 0, stream>>>(preQ, qcW, qcB, actQ, CQ);
    conv_silu_tr<<<dim3((unsigned)((n + 255) / 256)), 256, 0, stream>>>(preK, kcW, kcB, actK, CQ);
    n = (size_t)CV * NTOK;
    conv_silu_tr<<<dim3((unsigned)((n + 255) / 256)), 256, 0, stream>>>(preV, vcW, vcB, actV, CV);
  }

  // 4) gates
  gates_kernel<<<dim3(NTOK), 256, 0, stream>>>(x, Wa, Wb, alB, beB);

  // 5) L2 norms of q, k (per 128-dim head vector)
  l2norm_kernel<<<dim3(NTOK * HKc), 128, 0, stream>>>(actQ);
  l2norm_kernel<<<dim3(NTOK * HKc), 128, 0, stream>>>(actK);

  // 6) serial recurrence -> bf16 outs + final state
  recurrence_kernel<<<dim3(BB * HKc), 256, 0, stream>>>(
      actQ, actK, actV, alB, beB, state, outs_bf, stateF);

  // 7) output projection (WMMA bf16): out[token][H]
  gemm_bf16_wmma<true><<<dim3(NTOK / 64, HH / 128), 256, 0, stream>>>(
      wo_bf, outs_bf, outF, HH, CV, NTOK);

  (void)in_sizes; (void)n_in; (void)out_size; (void)ws_size;
}
